// EFNtoLocal_10943576670836
// MI455X (gfx1250) — compile-verified, matrix operand order checked by reading, not run on hardware
//
#include <hip/hip_runtime.h>

// ---------------- problem constants ----------------
#define N_NODES 50000
#define N_EDGES 800000
#define D_X     128
#define D_S     16
#define D_NODE  144        // D_X + D_S
#define D_IN    288        // 2 * D_NODE
#define D_H     128
#define D_OUT   128

#define KC1     9          // 288 / 32 K-chunks for GEMM1
#define KC2     4          // 128 / 32 K-chunks for GEMM2
#define NT      8          // 128 / 16 N-tiles
#define TILES   4          // 16-edge M-tiles per block
#define EPB     (TILES*16) // 64 edges per block
#define NBLK    (N_EDGES / EPB)   // 12500

// padded LDS row widths (bf16 elems); +8 keeps 16B alignment, breaks bank aliasing
#define SA_W    (D_IN + 8)   // 296
#define SH_W    (D_H + 8)    // 136

typedef __bf16 bf16_t;
typedef __attribute__((ext_vector_type(16))) __bf16 v16bf;
typedef __attribute__((ext_vector_type(8)))  float  v8f;

union Frag { v16bf v; uint4 q[2]; };

// workspace layout (bytes)
#define H_BYTES   ((size_t)N_NODES * D_NODE * 2)          // 14,400,000 (16B aligned)
#define W1P_ELEMS (KC1 * NT * 512)                        // 72 frags * 512 bf16
#define W2P_ELEMS (KC2 * NT * 512)                        // 32 frags * 512 bf16
#define W1P_BYTES ((size_t)W1P_ELEMS * 2)

// ---------------- kernel 1: node feature table (bf16) ----------------
__global__ void build_h_kernel(const float* __restrict__ x,
                               const float* __restrict__ scalars,
                               const int*   __restrict__ batch,
                               bf16_t*      __restrict__ h) {
    long long idx = (long long)blockIdx.x * blockDim.x + threadIdx.x;
    if (idx >= (long long)N_NODES * D_NODE) return;
    int n = (int)(idx / D_NODE);
    int c = (int)(idx % D_NODE);
    float v = (c < D_X) ? x[(long long)n * D_X + c]
                        : scalars[(long long)batch[n] * D_S + (c - D_X)];
    h[idx] = (bf16_t)v;
}

// ---------------- kernel 2: pack W1/W2 into WMMA B-fragment order ----------------
// fragment f = kc*NT + ntile; per fragment: 32 lanes x 16 bf16 (32B/lane, contiguous)
// element j of lane l -> B[kc*32 + kmap(j, l/16)][ntile*16 + l%16]
__device__ __forceinline__ int kmap(int j, int hi) {
    return (j < 8) ? (hi * 8 + j) : (16 + hi * 8 + (j - 8));
}

__global__ void pack_weights_kernel(const float* __restrict__ W1,
                                    const float* __restrict__ W2,
                                    bf16_t* __restrict__ w1p,
                                    bf16_t* __restrict__ w2p) {
    int t = blockIdx.x * blockDim.x + threadIdx.x;
    if (t < W1P_ELEMS) {
        int f = t >> 9, l = (t >> 4) & 31, j = t & 15;
        int kc = f >> 3, nt = f & 7;
        int n = (nt << 4) + (l & 15);
        int k = (kc << 5) + kmap(j, l >> 4);
        w1p[t] = (bf16_t)W1[k * D_H + n];
    } else if (t < W1P_ELEMS + W2P_ELEMS) {
        int u = t - W1P_ELEMS;
        int f = u >> 9, l = (u >> 4) & 31, j = u & 15;
        int kc = f >> 3, nt = f & 7;
        int n = (nt << 4) + (l & 15);
        int k = (kc << 5) + kmap(j, l >> 4);
        w2p[u] = (bf16_t)W2[k * D_OUT + n];
    }
}

// ---------------- kernel 3: zero the scatter accumulator ----------------
__global__ void zero_out_kernel(float4* __restrict__ out4) {
    long long i = (long long)blockIdx.x * blockDim.x + threadIdx.x;
    if (i < (long long)N_NODES * D_OUT / 4) out4[i] = float4{0.f, 0.f, 0.f, 0.f};
}

// ---------------- kernel 4: edge MLP + scatter-add ----------------
__launch_bounds__(256)
__global__ void ptconv_kernel(const bf16_t* __restrict__ h,
                              const bf16_t* __restrict__ w1p,
                              const bf16_t* __restrict__ w2p,
                              const float*  __restrict__ b1,
                              const float*  __restrict__ b2,
                              const int*    __restrict__ edge_index,
                              float*        __restrict__ out) {
    __shared__ __align__(16) bf16_t sA[EPB][SA_W];  // msg_in tile, bf16
    __shared__ __align__(16) bf16_t sH[EPB][SH_W];  // hidden tile, bf16
    __shared__ int sSrc[EPB];
    __shared__ int sDst[EPB];

    const int tid   = threadIdx.x;
    const int lane  = tid & 31;
    const int ntile = tid >> 5;     // wave id == output N-tile, 0..7
    const int ln    = lane & 15;    // row (A) or column (B/C/D) within tile
    const int hi    = lane >> 4;    // lane half selects K sub-range / M offset

    const long long edge_base = (long long)blockIdx.x * EPB;

    // ---- cache edge endpoints ----
    if (tid < EPB) {
        long long e = edge_base + tid;
        sSrc[tid] = edge_index[e];            // row 0: j (source)
        sDst[tid] = edge_index[N_EDGES + e];  // row 1: i (target)
    }
    __syncthreads();

    // ---- stage msg_in = [h[dst] | h[src]] : 64 rows x 144 u32 words ----
    {
        const int WPR = D_IN / 2;  // 144 u32 per row
        for (int idx = tid; idx < EPB * WPR; idx += 256) {
            int m = idx / WPR;
            int w = idx - m * WPR;
            int node = (w < 72) ? sDst[m] : sSrc[m];
            const unsigned int* hrow =
                (const unsigned int*)(h + (long long)node * D_NODE);
            ((unsigned int*)&sA[m][0])[w] = hrow[(w < 72) ? w : (w - 72)];
        }
    }
    __syncthreads();

    // ---- GEMM1: [64 x 288] @ W1[288 x 128] -> hidden, wave owns 16 cols ----
    v8f acc[TILES];
    {
        float bv = b1[ntile * 16 + ln];
        for (int t = 0; t < TILES; ++t)
            for (int r = 0; r < 8; ++r) acc[t][r] = bv;
    }
    for (int kc = 0; kc < KC1; ++kc) {
        Frag fb;
        const uint4* wp = ((const uint4*)w1p) +
                          ((size_t)(kc * NT + ntile) * 32 + lane) * 2;
        fb.q[0] = wp[0];
        fb.q[1] = wp[1];
        const int k0 = kc * 32 + hi * 8;
        for (int t = 0; t < TILES; ++t) {
            Frag fa;
            const bf16_t* row = &sA[t * 16 + ln][0];
            fa.q[0] = *(const uint4*)(row + k0);
            fa.q[1] = *(const uint4*)(row + k0 + 16);
            acc[t] = __builtin_amdgcn_wmma_f32_16x16x32_bf16(
                false, fa.v, false, fb.v, (short)0, acc[t], false, false);
        }
    }

    // ---- ReLU -> bf16 hidden tile in LDS ----
    for (int t = 0; t < TILES; ++t)
        for (int r = 0; r < 8; ++r) {
            int m = t * 16 + r + 8 * hi;   // C layout: lanes 0-15 M=r, 16-31 M=r+8
            float v = acc[t][r];
            sH[m][ntile * 16 + ln] = (bf16_t)(v > 0.f ? v : 0.f);
        }
    __syncthreads();

    // ---- GEMM2: hidden[64 x 128] @ W2[128 x 128], bias b2 in accumulator ----
    {
        float bv = b2[ntile * 16 + ln];
        for (int t = 0; t < TILES; ++t)
            for (int r = 0; r < 8; ++r) acc[t][r] = bv;
    }
    for (int kc = 0; kc < KC2; ++kc) {
        Frag fb;
        const uint4* wp = ((const uint4*)w2p) +
                          ((size_t)(kc * NT + ntile) * 32 + lane) * 2;
        fb.q[0] = wp[0];
        fb.q[1] = wp[1];
        const int k0 = kc * 32 + hi * 8;
        for (int t = 0; t < TILES; ++t) {
            Frag fa;
            const bf16_t* row = &sH[t * 16 + ln][0];
            fa.q[0] = *(const uint4*)(row + k0);
            fa.q[1] = *(const uint4*)(row + k0 + 16);
            acc[t] = __builtin_amdgcn_wmma_f32_16x16x32_bf16(
                false, fa.v, false, fb.v, (short)0, acc[t], false, false);
        }
    }

    // ---- scatter-add per destination node (native f32 atomics) ----
    for (int t = 0; t < TILES; ++t)
        for (int r = 0; r < 8; ++r) {
            int m = t * 16 + r + 8 * hi;
            int node = sDst[m];
            float* p = out + (long long)node * D_OUT + ntile * 16 + ln;
            __hip_atomic_fetch_add(p, acc[t][r], __ATOMIC_RELAXED,
                                   __HIP_MEMORY_SCOPE_AGENT);
        }
}

// ---------------- host launcher ----------------
extern "C" void kernel_launch(void* const* d_in, const int* in_sizes, int n_in,
                              void* d_out, int out_size, void* d_ws, size_t ws_size,
                              hipStream_t stream) {
    (void)in_sizes; (void)n_in; (void)out_size; (void)ws_size;

    const float* x       = (const float*)d_in[0];
    const float* scalars = (const float*)d_in[1];
    const float* W1      = (const float*)d_in[2];
    const float* b1      = (const float*)d_in[3];
    const float* W2      = (const float*)d_in[4];
    const float* b2      = (const float*)d_in[5];
    const int*   eidx    = (const int*)d_in[6];
    const int*   batch   = (const int*)d_in[7];
    float*       out     = (float*)d_out;

    bf16_t* h   = (bf16_t*)d_ws;
    bf16_t* w1p = (bf16_t*)((char*)d_ws + H_BYTES);
    bf16_t* w2p = (bf16_t*)((char*)d_ws + H_BYTES + W1P_BYTES);

    {   // node feature table
        long long total = (long long)N_NODES * D_NODE;
        int blocks = (int)((total + 255) / 256);
        build_h_kernel<<<blocks, 256, 0, stream>>>(x, scalars, batch, h);
    }
    {   // weight fragment packing
        int total = W1P_ELEMS + W2P_ELEMS;
        int blocks = (total + 255) / 256;
        pack_weights_kernel<<<blocks, 256, 0, stream>>>(W1, W2, w1p, w2p);
    }
    {   // zero accumulator (must happen every replay: atomics accumulate)
        long long total = (long long)N_NODES * D_OUT / 4;
        int blocks = (int)((total + 255) / 256);
        zero_out_kernel<<<blocks, 256, 0, stream>>>((float4*)out);
    }
    {   // main edge-MLP + scatter
        ptconv_kernel<<<NBLK, 256, 0, stream>>>(h, w1p, w2p, b1, b2, eidx, out);
    }
}
